// BiLSTM_CRF_26946624815751
// MI455X (gfx1250) — compile-verified, matrix operand order checked by reading, not run on hardware
//
#include <hip/hip_runtime.h>
#include <hip/hip_bf16.h>
#include <math.h>

// ---------------- problem constants ----------------
#define VOCAB 32000
#define E_DIM 256
#define HID   512
#define H_DIM 256          // per-direction hidden
#define G4    1024         // 4*H
#define KTAG  9
#define BB    32
#define TT    512
#define BT    (BB*TT)      // 16384

typedef __attribute__((ext_vector_type(16))) __bf16 bf16x16;
typedef __attribute__((ext_vector_type(8)))  float  v8f;

// k-offset inside a 32-wide K step for element e (0..15) of a 16-bit WMMA
// A/B fragment, given lane (ISA 7.12.2: lanes 0-15 K={0..7,16..23}, lanes
// 16-31 K={8..15,24..31}, 2 elements per VGPR).
__device__ __forceinline__ int kmap16(int e, int lane) {
  int r = e >> 1;
  int base = (r < 4) ? (r << 1) : (16 + ((r - 4) << 1));
  return base + (e & 1) + ((lane >= 16) ? 8 : 0);
}

__device__ __forceinline__ float sigf(float x) { return 1.0f / (1.0f + __expf(-x)); }

// ---------------- kernel 1: embedding gather -> bf16 ----------------
__global__ void gather_kernel(const int* __restrict__ ids,
                              const float* __restrict__ emb,
                              __bf16* __restrict__ xbf) {
  int row = blockIdx.x;            // 0..BT-1
  int e = threadIdx.x;             // 0..255
  int id = ids[row];
  xbf[row * E_DIM + e] = (__bf16)emb[(size_t)id * E_DIM + e];
}

// ---------------- kernel 2: input projection GEMM (both dirs) ----------------
// XP[dir] = Xbf(16384x256) @ w_ih[dir]^T (256x1024) + (b_ih+b_hh), f32 out.
__global__ void proj_kernel(const __bf16* __restrict__ xbf,
                            const float* __restrict__ w_ih_f,
                            const float* __restrict__ w_ih_b,
                            const float* __restrict__ b_ih_f,
                            const float* __restrict__ b_hh_f,
                            const float* __restrict__ b_ih_b,
                            const float* __restrict__ b_hh_b,
                            float* __restrict__ xp) {        // [2][BT][G4]
  __shared__ __align__(32) __bf16 sB[4 * 8 * 32 * 16];       // 32 KB: 4 n-tiles x 8 k-steps
  const int tid = threadIdx.x;
  const int lane = tid & 31;
  const int wave = tid >> 5;                                  // 0..7
  const int dir = blockIdx.z;
  const int n0blk = blockIdx.x * 64;                          // N block
  const int m0blk = blockIdx.y * 64;                          // M block
  const float* wih = dir ? w_ih_b : w_ih_f;

  // stage B fragments (bf16, WMMA lane layout) for this 64-column block
  for (int rec = tid; rec < 4 * 8 * 32; rec += 256) {
    int l  = rec & 31;
    int ks = (rec >> 5) & 7;
    int nt = rec >> 8;
    int col = n0blk + nt * 16 + (l & 15);                     // gate column
    __bf16* dst = &sB[rec * 16];
#pragma unroll
    for (int e = 0; e < 16; e++) {
      int k = ks * 32 + kmap16(e, l);
      dst[e] = (__bf16)wih[col * E_DIM + k];
    }
  }
  __syncthreads();

  const int mt  = wave & 3;                                   // m-tile 0..3
  const int ntp = (wave >> 2) * 2;                            // n-tiles ntp, ntp+1
  const int row = m0blk + mt * 16 + (lane & 15);
  const int khi = (lane >= 16) ? 8 : 0;

  v8f acc0 = {};
  v8f acc1 = {};
#pragma unroll
  for (int ks = 0; ks < 8; ks++) {
    union { unsigned int u[8]; bf16x16 v; } A;
#pragma unroll
    for (int r = 0; r < 4; r++) {
      int k = ks * 32 + 2 * r + khi;
      A.u[r] = *(const unsigned int*)(xbf + (size_t)row * E_DIM + k);
    }
#pragma unroll
    for (int r = 4; r < 8; r++) {
      int k = ks * 32 + 16 + 2 * (r - 4) + khi;
      A.u[r] = *(const unsigned int*)(xbf + (size_t)row * E_DIM + k);
    }
    bf16x16 b0 = *(const bf16x16*)&sB[((ntp * 8 + ks) * 32 + lane) * 16];
    bf16x16 b1 = *(const bf16x16*)&sB[(((ntp + 1) * 8 + ks) * 32 + lane) * 16];
    acc0 = __builtin_amdgcn_wmma_f32_16x16x32_bf16(false, A.v, false, b0, (short)0, acc0, false, false);
    acc1 = __builtin_amdgcn_wmma_f32_16x16x32_bf16(false, A.v, false, b1, (short)0, acc1, false, false);
  }

  const float* bih = dir ? b_ih_b : b_ih_f;
  const float* bhh = dir ? b_hh_b : b_hh_f;
  float* xpd = xp + (size_t)dir * BT * G4;
  int mrow0 = m0blk + mt * 16 + 8 * (lane >> 4);
  {
    int col = n0blk + ntp * 16 + (lane & 15);
    float bv = bih[col] + bhh[col];
#pragma unroll
    for (int r = 0; r < 8; r++) xpd[(size_t)(mrow0 + r) * G4 + col] = acc0[r] + bv;
  }
  {
    int col = n0blk + (ntp + 1) * 16 + (lane & 15);
    float bv = bih[col] + bhh[col];
#pragma unroll
    for (int r = 0; r < 8; r++) xpd[(size_t)(mrow0 + r) * G4 + col] = acc1[r] + bv;
  }
}

// ---------------- kernel 3: persistent BiLSTM recurrence ----------------
// 8 workgroups: dir = blk/4 (0=fwd,1=bwd); each WG owns 64 h-columns
// (=> 256 gate columns i/f/g/o) with its w_hh slice resident in LDS as
// pre-swizzled bf16 WMMA fragments. Per step: stage h -> A-frags (LDS),
// 16 v_wmma per wave, gates to LDS, fused sigmoid/tanh cell update,
// h exchanged via L2 with a release/acquire spin barrier.
__global__ void __launch_bounds__(512, 1)
lstm_kernel(const float* __restrict__ xp,        // [2][BT][G4]
            const float* __restrict__ w_hh_f,
            const float* __restrict__ w_hh_b,
            float* __restrict__ hbuf,            // [2 dir][2 phase][B][H]
            float* __restrict__ lstm_out,        // [BT][HID]
            int* __restrict__ sync) {            // [2]
  // LDS partition: B-frags 128KB | A-frags 16KB | gates 33KB  (~177KB)
  __shared__ __align__(32) char smem[16*8*32*16*2 + 2*8*32*16*2 + 256*33*4];
  __bf16* sB = (__bf16*)smem;                         // [16][8][32][16]
  __bf16* sA = (__bf16*)(smem + 16*8*32*16*2);        // [2][8][32][16]
  float*  sG = (float*)(smem + 16*8*32*16*2 + 2*8*32*16*2); // [256][33] padded

  const int tid  = threadIdx.x;
  const int lane = tid & 31;
  const int wave = tid >> 5;                           // 0..15 = n-tile
  const int dir   = blockIdx.x >> 2;
  const int slice = blockIdx.x & 3;
  const int j0 = slice * 64;                           // h-column block
  const float* whh = dir ? w_hh_b : w_hh_f;
  const float* xpd = xp + (size_t)dir * BT * G4;

  // ---- one-time: swizzle this WG's w_hh slice into bf16 B fragments ----
  for (int rec = tid; rec < 16 * 8 * 32; rec += 512) {
    int l  = rec & 31;
    int ks = (rec >> 5) & 7;
    int nt = rec >> 8;
    int s  = nt * 16 + (l & 15);                       // slice col 0..255
    int q  = s >> 6;                                   // gate 0..3
    int jj = s & 63;
    int row = q * H_DIM + j0 + jj;                     // w_hh row (gate unit)
    __bf16* dst = &sB[rec * 16];
#pragma unroll
    for (int e = 0; e < 16; e++) {
      int k = ks * 32 + kmap16(e, l);
      dst[e] = (__bf16)whh[(size_t)row * H_DIM + k];
    }
  }
  __syncthreads();

  // persistent per-thread cell state: 4 (b, jj) pairs each
  float creg[4] = {0.f, 0.f, 0.f, 0.f};

  for (int t = 0; t < TT; t++) {
    const int tt = dir ? (TT - 1 - t) : t;             // time index in xp / output
    const float* hp = hbuf + (size_t)(dir * 2 + (t & 1)) * BB * H_DIM;
    float* hn = hbuf + (size_t)(dir * 2 + ((t + 1) & 1)) * BB * H_DIM;

    // ---- stage h_prev -> bf16 A fragments (exactly one record per thread) ----
    {
      int l  = tid & 31;
      int ks = (tid >> 5) & 7;
      int mt = tid >> 8;                               // 0..1
      int m = mt * 16 + (l & 15);
      __bf16* dst = &sA[tid * 16];
#pragma unroll
      for (int e = 0; e < 16; e++) {
        int k = ks * 32 + kmap16(e, l);
        dst[e] = (__bf16)hp[m * H_DIM + k];
      }
    }
    __syncthreads();

    // ---- GEMM: gates(32 x 256slice) += h(32x256) @ Whh_slice^T ----
    v8f acc0 = {};
    v8f acc1 = {};
#pragma unroll
    for (int ks = 0; ks < 8; ks++) {
      bf16x16 bfr = *(const bf16x16*)&sB[((wave * 8 + ks) * 32 + lane) * 16];
      bf16x16 a0  = *(const bf16x16*)&sA[((ks) * 32 + lane) * 16];
      bf16x16 a1  = *(const bf16x16*)&sA[((8 + ks) * 32 + lane) * 16];
      acc0 = __builtin_amdgcn_wmma_f32_16x16x32_bf16(false, a0, false, bfr, (short)0, acc0, false, false);
      acc1 = __builtin_amdgcn_wmma_f32_16x16x32_bf16(false, a1, false, bfr, (short)0, acc1, false, false);
    }
    {
      int s  = wave * 16 + (lane & 15);
      int mb = 8 * (lane >> 4);
#pragma unroll
      for (int r = 0; r < 8; r++) sG[s * 33 + mb + r]      = acc0[r];
#pragma unroll
      for (int r = 0; r < 8; r++) sG[s * 33 + 16 + mb + r] = acc1[r];
    }
    __syncthreads();

    // ---- fused cell update for this WG's 64 h-columns ----
#pragma unroll
    for (int p = 0; p < 4; p++) {
      int pidx = tid + p * 512;                        // 0..2047
      int b  = pidx >> 6;
      int jj = pidx & 63;
      size_t xrow = ((size_t)b * TT + tt) * G4;
      float gi = sG[(0 * 64 + jj) * 33 + b] + xpd[xrow + 0 * H_DIM + j0 + jj];
      float gf = sG[(1 * 64 + jj) * 33 + b] + xpd[xrow + 1 * H_DIM + j0 + jj];
      float gg = sG[(2 * 64 + jj) * 33 + b] + xpd[xrow + 2 * H_DIM + j0 + jj];
      float go = sG[(3 * 64 + jj) * 33 + b] + xpd[xrow + 3 * H_DIM + j0 + jj];
      float c = sigf(gf) * creg[p] + sigf(gi) * tanhf(gg);
      creg[p] = c;
      float hv = sigf(go) * tanhf(c);
      hn[b * H_DIM + j0 + jj] = hv;
      lstm_out[((size_t)b * TT + tt) * HID + dir * H_DIM + j0 + jj] = hv;
    }

    // ---- inter-WG step barrier (4 WGs per direction) ----
    __threadfence();
    __syncthreads();
    if (tid == 0) {
      __hip_atomic_fetch_add(&sync[dir], 1, __ATOMIC_RELEASE, __HIP_MEMORY_SCOPE_AGENT);
      while (__hip_atomic_load(&sync[dir], __ATOMIC_ACQUIRE, __HIP_MEMORY_SCOPE_AGENT) < 4 * (t + 1)) {
        __builtin_amdgcn_s_sleep(2);
      }
    }
    __syncthreads();
  }
}

// ---------------- kernel 4: emissions (N=9 too small for WMMA) ----------------
__global__ void emis_kernel(const float* __restrict__ lstm_out,
                            const float* __restrict__ w_out,
                            const float* __restrict__ b_out,
                            float* __restrict__ emis) {
  int lane = threadIdx.x & 31;
  int wave = threadIdx.x >> 5;
  int row = blockIdx.x * 8 + wave;                     // 0..BT-1
  const float* h = lstm_out + (size_t)row * HID;
  for (int k = 0; k < KTAG; k++) {
    float s = 0.f;
    for (int i = lane; i < HID; i += 32) s += h[i] * w_out[k * HID + i];
#pragma unroll
    for (int off = 16; off > 0; off >>= 1) s += __shfl_xor(s, off, 32);
    if (lane == 0) emis[(size_t)row * KTAG + k] = s + b_out[k];
  }
}

// ---------------- kernel 5: CRF NLL (wave per batch) ----------------
__global__ void crf_kernel(const float* __restrict__ emis,
                           const int* __restrict__ tags,
                           const unsigned char* __restrict__ mask,
                           const float* __restrict__ trans,
                           const float* __restrict__ startt,
                           const float* __restrict__ endt,
                           float* __restrict__ out) {
  __shared__ float tr[KTAG * KTAG];
  __shared__ float llh[BB];
  int tid = threadIdx.x;
  if (tid < KTAG * KTAG) tr[tid] = trans[tid];
  __syncthreads();

  int lane = tid & 31;
  int b = tid >> 5;                                    // batch = wave
  const float* em = emis + (size_t)b * TT * KTAG;
  const int* tg = tags + (size_t)b * TT;
  const unsigned char* mk = mask + (size_t)b * TT;

  // gold-path score (lane-parallel over t)
  float sc = 0.f;
  int cnt = 0;
  for (int t = lane; t < TT; t += 32) cnt += (mk[t] != 0);
  for (int t = lane + 1; t < TT; t += 32)
    if (mk[t]) sc += tr[tg[t - 1] * KTAG + tg[t]] + em[t * KTAG + tg[t]];
#pragma unroll
  for (int off = 16; off > 0; off >>= 1) {
    sc  += __shfl_xor(sc, off, 32);
    cnt += __shfl_xor(cnt, off, 32);
  }
  sc += startt[tg[0]] + em[tg[0]] + endt[tg[cnt - 1]];

  // forward algorithm: lane j (<9) carries alpha[j]
  const int jcl = (lane < KTAG) ? lane : 0;
  float alpha = (lane < KTAG) ? (startt[jcl] + em[jcl]) : -1e30f;
  for (int t = 1; t < TT; t++) {
    float av[KTAG];
#pragma unroll
    for (int i = 0; i < KTAG; i++) av[i] = __shfl(alpha, i, 32);
    float m = -1e30f;
#pragma unroll
    for (int i = 0; i < KTAG; i++) m = fmaxf(m, av[i] + tr[i * KTAG + jcl]);
    float se = 0.f;
#pragma unroll
    for (int i = 0; i < KTAG; i++) se += __expf(av[i] + tr[i * KTAG + jcl] - m);
    float nxt = m + __logf(se) + em[t * KTAG + jcl];
    if (lane < KTAG && mk[t]) alpha = nxt;
  }
  float v = (lane < KTAG) ? (alpha + endt[jcl]) : -1e30f;
  float mm = v;
#pragma unroll
  for (int off = 16; off > 0; off >>= 1) mm = fmaxf(mm, __shfl_xor(mm, off, 32));
  float se = (lane < KTAG) ? __expf(v - mm) : 0.f;
#pragma unroll
  for (int off = 16; off > 0; off >>= 1) se += __shfl_xor(se, off, 32);
  float logZ = mm + __logf(se);
  if (lane == 0) llh[b] = sc - logZ;
  __syncthreads();
  if (tid == 0) {
    float acc = 0.f;
    for (int i = 0; i < BB; i++) acc += llh[i];
    out[0] = -acc / (float)BB;
  }
}

// ---------------- host: workspace layout + launch sequence ----------------
extern "C" void kernel_launch(void* const* d_in, const int* in_sizes, int n_in,
                              void* d_out, int out_size, void* d_ws, size_t ws_size,
                              hipStream_t stream) {
  const int*   input_ids = (const int*)d_in[0];
  const int*   tags      = (const int*)d_in[1];
  const unsigned char* mask = (const unsigned char*)d_in[2];
  const float* emb    = (const float*)d_in[3];
  const float* w_ih_f = (const float*)d_in[4];
  const float* w_hh_f = (const float*)d_in[5];
  const float* b_ih_f = (const float*)d_in[6];
  const float* b_hh_f = (const float*)d_in[7];
  const float* w_ih_b = (const float*)d_in[8];
  const float* w_hh_b = (const float*)d_in[9];
  const float* b_ih_b = (const float*)d_in[10];
  const float* b_hh_b = (const float*)d_in[11];
  const float* w_out  = (const float*)d_in[12];
  const float* b_out  = (const float*)d_in[13];
  const float* trans  = (const float*)d_in[14];
  const float* startt = (const float*)d_in[15];
  const float* endt   = (const float*)d_in[16];

  char* ws = (char*)d_ws;
  size_t off = 0;
  __bf16* xbf = (__bf16*)(ws + off);      off += (size_t)BT * E_DIM * 2;        // 8 MB
  float* xp   = (float*)(ws + off);       off += (size_t)2 * BT * G4 * 4;       // 128 MB
  float* lstm_out = (float*)(ws + off);   off += (size_t)BT * HID * 4;          // 32 MB
  float* emis = (float*)(ws + off);       off += (size_t)BT * KTAG * 4;         // 576 KB
  float* hbuf = (float*)(ws + off);       size_t hbytes = (size_t)2 * 2 * BB * H_DIM * 4;
  off += hbytes;
  int* syncp  = (int*)(ws + off);         off += 256;

  // h state + inter-WG sync counters must start at zero every call
  hipMemsetAsync(hbuf, 0, hbytes + 256, stream);

  gather_kernel<<<BT, E_DIM, 0, stream>>>(input_ids, emb, xbf);

  proj_kernel<<<dim3(G4 / 64, BT / 64, 2), 256, 0, stream>>>(
      xbf, w_ih_f, w_ih_b, b_ih_f, b_hh_f, b_ih_b, b_hh_b, xp);

  lstm_kernel<<<8, 512, 0, stream>>>(xp, w_hh_f, w_hh_b, hbuf, lstm_out, syncp);

  emis_kernel<<<BT / 8, 256, 0, stream>>>(lstm_out, w_out, b_out, emis);

  crf_kernel<<<1, BB * 32, 0, stream>>>(emis, tags, mask, trans, startt, endt,
                                        (float*)d_out);
}
